// LSTMDecoder_12584254178071
// MI455X (gfx1250) — matrix operand order repaired and checked by reading, not compiled, not run on hardware
//
#include <hip/hip_runtime.h>
#include <hip/hip_bf16.h>
#include <stdint.h>

// Problem constants
#define BATCH  8192
#define HID    512
#define OUTD   80
#define OUTP   96      // padded hidden size for layer 2 (80 -> 96, multiple of 32)
#define TSTEPS 32

// LDS staging geometry: 4 gate tiles of 16 rows x 512 bf16, row padded by 16B
#define LDS_ROW_BYTES  1040                   // 1024 data + 16 pad (4 dwords)
#define LDS_GATE_BYTES (16 * LDS_ROW_BYTES)   // 16,640
#define LDS_TOTAL      (4 * LDS_GATE_BYTES)   // 66,560 (<320KB/WGP)

typedef __bf16 bf16_t;
typedef __attribute__((ext_vector_type(16))) __bf16 bf16x16;
typedef __attribute__((ext_vector_type(8)))  float  f32x8;
typedef __attribute__((ext_vector_type(4)))  unsigned int v4u;
typedef __attribute__((ext_vector_type(8)))  int v8i;
typedef __attribute__((ext_vector_type(4)))  int v4i;

union FragU { bf16x16 v; uint4 q[2]; };

// ---------------------------------------------------------------------------
// TDM: stage one 16-row x K(=512) bf16 weight tile into LDS at lds_addr.
// D# packed per CDNA5 ISA ch.8: 2D tile, data_size=2B, LDS padding of
// 4 dwords every 256 dwords (one 1024B row) -> 1040B LDS row stride.
// Descriptor operands are workgroup-uniform (blockIdx / kernel args only).
// ---------------------------------------------------------------------------
__device__ __forceinline__ void tdm_stage_tile(const bf16_t* gptr, unsigned int lds_addr) {
  const uint64_t ga = (uint64_t)(uintptr_t)gptr;
  // group0: count=1 | lds_addr | global_addr | type=2
  v4u g0 = { 1u,
             lds_addr,
             (unsigned int)(ga & 0xFFFFFFFFu),
             (unsigned int)((ga >> 32) & 0x01FFFFFFu) | (2u << 30) };
  // group1: data_size=1(2B), pad_enable, pad_interval=7(256dw), pad_amount=3(4dw)
  const unsigned int w0 = (1u << 16) | (1u << 20) | (7u << 22) | (3u << 25);
  const unsigned int td0 = HID;     // tensor_dim0 = 512 elements per row
  const unsigned int td1 = 16u;     // tensor_dim1 = 16 rows (== tile, no OOB)
  v8i g1 = { (int)w0,
             (int)((td0 & 0xFFFFu) << 16),             // [47:32]=0, [63:48]=td0.lo
             (int)((td0 >> 16) | (td1 << 16)),         // td0.hi | td1.lo
             (int)((td1 >> 16) | (HID << 16)),         // td1.hi | tile_dim0=512
             (int)16,                                  // tile_dim1=16, tile_dim2=0
             (int)HID,                                 // tensor_dim0_stride.lo = 512
             0, 0 };                                   // stride.hi, dim1_stride=0
  v4i z4 = { 0, 0, 0, 0 };
  v8i z8 = { 0, 0, 0, 0, 0, 0, 0, 0 };
  __builtin_amdgcn_tensor_load_to_lds(g0, g1, z4, z4, z8, 0);
}

// ---------------------------------------------------------------------------
// Opaque LDS B-fragment load: the TDM builtin carries no pointer arg, so the
// compiler cannot see that smW was written -- plain C++ loads get folded to
// undef.  Issue the two 16B LDS reads as inline asm with an explicit DScnt
// wait instead (addr = raw 32-bit LDS byte address).
// ---------------------------------------------------------------------------
__device__ __forceinline__ bf16x16 lds_b_frag(unsigned int addr) {
  uint4 q0, q1;
  asm volatile("ds_load_b128 %0, %2\n\t"
               "ds_load_b128 %1, %2 offset:16\n\t"
               "s_wait_dscnt 0x0"
               : "=v"(q0), "=v"(q1)
               : "v"(addr));
  FragU f;
  f.q[0] = q0;
  f.q[1] = q1;
  return f.v;
}

// ---------------------------------------------------------------------------
// Fragment loaders for wave32 v_wmma_f32_16x16x32_bf16 (contiguous k order used here, same for A and B)
// A (16x32): lane m<16 holds row m K=0..15; lane m+16 holds row m K=16..31.
// B (32x16): lane n<16 holds col n K=0..15; lane n+16 holds col n K=16..31.
// ---------------------------------------------------------------------------
__device__ __forceinline__ bf16x16 load_a_frag(const bf16_t* __restrict__ A,
                                               int lda, int row0, int k0, int lane) {
  const int m  = row0 + (lane & 15);
  const int hi = lane >> 4;
  const bf16_t* p = A + (size_t)m * lda + (k0 + hi * 16);
  FragU f;
  f.q[0] = *(const uint4*)(p);
  f.q[1] = *(const uint4*)(p + 8);
  return f.v;
}

__device__ __forceinline__ bf16x16 load_b_frag_g(const bf16_t* __restrict__ W,
                                                 int ldw, int r0, int k0, int lane) {
  const int r  = r0 + (lane & 15);
  const int hi = lane >> 4;
  const bf16_t* p = W + (size_t)r * ldw + (k0 + hi * 16);
  FragU f;
  f.q[0] = *(const uint4*)(p);
  f.q[1] = *(const uint4*)(p + 8);
  return f.v;
}

__device__ __forceinline__ float fsigmoid(float x) {
  return 1.0f / (1.0f + __expf(-x));
}
__device__ __forceinline__ float ftanh_fast(float x) {
  return 2.0f / (1.0f + __expf(-2.0f * x)) - 1.0f;
}

// ---------------------------------------------------------------------------
// Fused LSTM layer.  Each wave owns TWO 16x16 (h,c) tiles (32 rows).
// WG = 8 waves = 256 rows x 16 cols.  Weights for the WG's 16 columns are
// TDM-staged in LDS (K1 path always; K2 path when K2==HID).
//   grid.x = BATCH/256, grid.y = Hd/16
// ---------------------------------------------------------------------------
__global__ __launch_bounds__(256)
void lstm_layer_wmma(const bf16_t* __restrict__ A1, const bf16_t* __restrict__ W1, int K1,
                     const bf16_t* __restrict__ A2, const bf16_t* __restrict__ W2, int K2,
                     int stage2,
                     const float*  __restrict__ bias,
                     float*        __restrict__ cstate,
                     bf16_t*       __restrict__ h_out, int ldh,
                     float*        __restrict__ y_out, int t, int Hd)
{
  __shared__ __align__(16) char smW[LDS_TOTAL];

  const int lane = threadIdx.x & 31;
  const int wave = threadIdx.x >> 5;
  const int row0 = (blockIdx.x * 8 + wave) * 32;   // two 16-row tiles per wave
  const int col0 = blockIdx.y * 16;
  const int nloc = lane & 15;
  const int hi   = lane >> 4;

  // Raw LDS byte address of smW (low 32 bits of the generic pointer are the
  // workgroup-relative LDS offset per the aperture mapping).
  const unsigned int smBase = (unsigned int)(uintptr_t)(&smW[0]);

  // Stage W1 gate tiles (rows g*Hd+col0 .. +15, full K=512) into LDS via TDM.
  if (wave == 0) {
#pragma unroll
    for (int g = 0; g < 4; ++g)
      tdm_stage_tile(W1 + (size_t)(g * Hd + col0) * K1,
                     smBase + (unsigned int)(g * LDS_GATE_BYTES));
    __builtin_amdgcn_s_wait_tensorcnt(0);
  }
  __syncthreads();

  // Accumulators initialized with the fused bias b_ih + b_hh (column-wise).
  f32x8 acc[2][4];
#pragma unroll
  for (int g = 0; g < 4; ++g) {
    const float bv = bias[g * Hd + col0 + nloc];
    f32x8 c0 = {bv, bv, bv, bv, bv, bv, bv, bv};
    acc[0][g] = c0;
    acc[1][g] = c0;
  }

  // Per-lane LDS fragment base (row stride 1040B avoids bank conflicts).
  const unsigned int smLane = smBase + (unsigned int)(nloc * LDS_ROW_BYTES + hi * 32);

  // ---- Input-path GEMM: B from LDS ----
  for (int k0 = 0; k0 < K1; k0 += 32) {
    const bf16x16 a0 = load_a_frag(A1, K1, row0,      k0, lane);
    const bf16x16 a1 = load_a_frag(A1, K1, row0 + 16, k0, lane);
#pragma unroll
    for (int g = 0; g < 4; ++g) {
      const bf16x16 b = lds_b_frag(smLane + (unsigned int)(g * LDS_GATE_BYTES + k0 * 2));
      acc[0][g] = __builtin_amdgcn_wmma_f32_16x16x32_bf16(
          false, a0, false, b, (short)0, acc[0][g], false, false);
      acc[1][g] = __builtin_amdgcn_wmma_f32_16x16x32_bf16(
          false, a1, false, b, (short)0, acc[1][g], false, false);
    }
  }

  // ---- Hidden-path GEMM ----
  if (A2) {
    if (stage2) {  // K2 == HID: restage LDS with W2 tiles
      __syncthreads();
      if (wave == 0) {
#pragma unroll
        for (int g = 0; g < 4; ++g)
          tdm_stage_tile(W2 + (size_t)(g * Hd + col0) * K2,
                         smBase + (unsigned int)(g * LDS_GATE_BYTES));
        __builtin_amdgcn_s_wait_tensorcnt(0);
      }
      __syncthreads();
      for (int k0 = 0; k0 < K2; k0 += 32) {
        const bf16x16 a0 = load_a_frag(A2, K2, row0,      k0, lane);
        const bf16x16 a1 = load_a_frag(A2, K2, row0 + 16, k0, lane);
#pragma unroll
        for (int g = 0; g < 4; ++g) {
          const bf16x16 b = lds_b_frag(smLane + (unsigned int)(g * LDS_GATE_BYTES + k0 * 2));
          acc[0][g] = __builtin_amdgcn_wmma_f32_16x16x32_bf16(
              false, a0, false, b, (short)0, acc[0][g], false, false);
          acc[1][g] = __builtin_amdgcn_wmma_f32_16x16x32_bf16(
              false, a1, false, b, (short)0, acc[1][g], false, false);
        }
      }
    } else {       // small K2 (layer 2, K=96): direct global B loads
      for (int k0 = 0; k0 < K2; k0 += 32) {
        const bf16x16 a0 = load_a_frag(A2, K2, row0,      k0, lane);
        const bf16x16 a1 = load_a_frag(A2, K2, row0 + 16, k0, lane);
#pragma unroll
        for (int g = 0; g < 4; ++g) {
          const bf16x16 b = load_b_frag_g(W2, K2, g * Hd + col0, k0, lane);
          acc[0][g] = __builtin_amdgcn_wmma_f32_16x16x32_bf16(
              false, a0, false, b, (short)0, acc[0][g], false, false);
          acc[1][g] = __builtin_amdgcn_wmma_f32_16x16x32_bf16(
              false, a1, false, b, (short)0, acc[1][g], false, false);
        }
      }
    }
  }

  // ---- Fused LSTM pointwise on C-layout: (M = v + 8*hi, N = lane&15) ----
  const int n = col0 + nloc;
#pragma unroll
  for (int mt = 0; mt < 2; ++mt) {
    const int rbase = row0 + mt * 16;
#pragma unroll
    for (int v = 0; v < 8; ++v) {
      const int m = rbase + v + hi * 8;
      const size_t cidx = (size_t)m * Hd + n;
      const float ig = fsigmoid(acc[mt][0][v]);
      const float fg = fsigmoid(acc[mt][1][v]);
      const float gg = ftanh_fast(acc[mt][2][v]);
      const float og = fsigmoid(acc[mt][3][v]);
      const float c_new = fg * cstate[cidx] + ig * gg;
      const float h_new = og * ftanh_fast(c_new);
      cstate[cidx] = c_new;
      h_out[(size_t)m * ldh + n] = (bf16_t)h_new;
      if (y_out) y_out[((size_t)m * TSTEPS + t) * OUTD + n] = h_new;  // [B,T,O]
    }
  }
}

// ---------------------------------------------------------------------------
// Prep kernels
// ---------------------------------------------------------------------------
__global__ void k_cvt_bf16(const float* __restrict__ s, bf16_t* __restrict__ d, int n) {
  int i = blockIdx.x * blockDim.x + threadIdx.x;
  if (i < n) d[i] = (bf16_t)s[i];
}
__global__ void k_sum2_bf16(const float* __restrict__ a, const float* __restrict__ b,
                            bf16_t* __restrict__ d, int n) {
  int i = blockIdx.x * blockDim.x + threadIdx.x;
  if (i < n) d[i] = (bf16_t)(a[i] + b[i]);
}
__global__ void k_sum2_f32(const float* __restrict__ a, const float* __restrict__ b,
                           float* __restrict__ d, int n) {
  int i = blockIdx.x * blockDim.x + threadIdx.x;
  if (i < n) d[i] = a[i] + b[i];
}
// w_hh2 [320,80] fp32 -> [320,96] bf16, zero padded
__global__ void k_pad_hh2(const float* __restrict__ s, bf16_t* __restrict__ d) {
  int i = blockIdx.x * blockDim.x + threadIdx.x;
  if (i < 4 * OUTD * OUTP) {
    int r = i / OUTP, k = i % OUTP;
    d[i] = (k < OUTD) ? (bf16_t)s[r * OUTD + k] : (bf16_t)0.0f;
  }
}
__global__ void k_zero_u32(uint32_t* __restrict__ p, int n) {
  int i = blockIdx.x * blockDim.x + threadIdx.x;
  if (i < n) p[i] = 0u;
}

// ---------------------------------------------------------------------------
// Orchestration
// ---------------------------------------------------------------------------
extern "C" void kernel_launch(void* const* d_in, const int* in_sizes, int n_in,
                              void* d_out, int out_size, void* d_ws, size_t ws_size,
                              hipStream_t stream) {
  (void)in_sizes; (void)n_in; (void)out_size; (void)ws_size;

  const float* x     = (const float*)d_in[0];
  const float* w_ih0 = (const float*)d_in[1];
  const float* w_hh0 = (const float*)d_in[2];
  const float* b_ih0 = (const float*)d_in[3];
  const float* b_hh0 = (const float*)d_in[4];
  const float* w_ih1 = (const float*)d_in[5];
  const float* w_hh1 = (const float*)d_in[6];
  const float* b_ih1 = (const float*)d_in[7];
  const float* b_hh1 = (const float*)d_in[8];
  const float* w_ih2 = (const float*)d_in[9];
  const float* w_hh2 = (const float*)d_in[10];
  const float* b_ih2 = (const float*)d_in[11];
  const float* b_hh2 = (const float*)d_in[12];
  float* out = (float*)d_out;

  char* ws = (char*)d_ws;
  size_t off = 0;
  auto alloc = [&](size_t bytes) -> void* {
    void* p = ws + off;
    off += (bytes + 255) & ~(size_t)255;
    return p;
  };

  const int WBIG = 4 * HID * HID;  // 2048*512
  bf16_t* wb_ih0  = (bf16_t*)alloc((size_t)WBIG * 2);
  bf16_t* wb_sum0 = (bf16_t*)alloc((size_t)WBIG * 2);   // w_ih0 + w_hh0 (t>=1 trick)
  bf16_t* wb_ih1  = (bf16_t*)alloc((size_t)WBIG * 2);
  bf16_t* wb_hh1  = (bf16_t*)alloc((size_t)WBIG * 2);
  bf16_t* wb_ih2  = (bf16_t*)alloc((size_t)4 * OUTD * HID * 2);
  bf16_t* wb_hh2p = (bf16_t*)alloc((size_t)4 * OUTD * OUTP * 2);
  float*  bias0   = (float*) alloc((size_t)4 * HID * 4);
  float*  bias1   = (float*) alloc((size_t)4 * HID * 4);
  float*  bias2   = (float*) alloc((size_t)4 * OUTD * 4);
  bf16_t* xb      = (bf16_t*)alloc((size_t)BATCH * HID * 2);
  bf16_t* h0b[2]  = { (bf16_t*)alloc((size_t)BATCH * HID * 2),
                      (bf16_t*)alloc((size_t)BATCH * HID * 2) };
  bf16_t* h1b[2]  = { (bf16_t*)alloc((size_t)BATCH * HID * 2),
                      (bf16_t*)alloc((size_t)BATCH * HID * 2) };
  bf16_t* h2b[2]  = { (bf16_t*)alloc((size_t)BATCH * OUTP * 2),
                      (bf16_t*)alloc((size_t)BATCH * OUTP * 2) };
  float*  c0      = (float*) alloc((size_t)BATCH * HID * 4);
  float*  c1      = (float*) alloc((size_t)BATCH * HID * 4);
  float*  c2      = (float*) alloc((size_t)BATCH * OUTD * 4);

  auto grid1d = [](int n) { return dim3((unsigned)((n + 255) / 256)); };

  // --- prep: bf16 weights, fused biases, zeroed state ---
  k_cvt_bf16 <<<grid1d(BATCH * HID), 256, 0, stream>>>(x, xb, BATCH * HID);
  k_cvt_bf16 <<<grid1d(WBIG), 256, 0, stream>>>(w_ih0, wb_ih0, WBIG);
  k_sum2_bf16<<<grid1d(WBIG), 256, 0, stream>>>(w_ih0, w_hh0, wb_sum0, WBIG);
  k_cvt_bf16 <<<grid1d(WBIG), 256, 0, stream>>>(w_ih1, wb_ih1, WBIG);
  k_cvt_bf16 <<<grid1d(WBIG), 256, 0, stream>>>(w_hh1, wb_hh1, WBIG);
  k_cvt_bf16 <<<grid1d(4 * OUTD * HID), 256, 0, stream>>>(w_ih2, wb_ih2, 4 * OUTD * HID);
  k_pad_hh2  <<<grid1d(4 * OUTD * OUTP), 256, 0, stream>>>(w_hh2, wb_hh2p);
  k_sum2_f32 <<<grid1d(4 * HID), 256, 0, stream>>>(b_ih0, b_hh0, bias0, 4 * HID);
  k_sum2_f32 <<<grid1d(4 * HID), 256, 0, stream>>>(b_ih1, b_hh1, bias1, 4 * HID);
  k_sum2_f32 <<<grid1d(4 * OUTD), 256, 0, stream>>>(b_ih2, b_hh2, bias2, 4 * OUTD);
  k_zero_u32 <<<grid1d(BATCH * HID), 256, 0, stream>>>((uint32_t*)c0, BATCH * HID);
  k_zero_u32 <<<grid1d(BATCH * HID), 256, 0, stream>>>((uint32_t*)c1, BATCH * HID);
  k_zero_u32 <<<grid1d(BATCH * OUTD), 256, 0, stream>>>((uint32_t*)c2, BATCH * OUTD);
  k_zero_u32 <<<grid1d(BATCH * OUTP / 2), 256, 0, stream>>>((uint32_t*)h2b[0], BATCH * OUTP / 2);
  k_zero_u32 <<<grid1d(BATCH * OUTP / 2), 256, 0, stream>>>((uint32_t*)h2b[1], BATCH * OUTP / 2);

  // --- recurrence: 3 fused GEMM+pointwise kernels per step ---
  const dim3 blk(256);
  const dim3 gridBig(BATCH / 256, HID / 16);   // 32 x 32 workgroups
  const dim3 gridOut(BATCH / 256, OUTD / 16);  // 32 x 5  workgroups

  int cur = 0;
  for (int t = 0; t < TSTEPS; ++t) {
    const int prev = cur ^ 1;

    // Layer 0: t==0 -> x @ w_ih0^T ; t>=1 -> h0_prev @ (w_ih0+w_hh0)^T
    const bf16_t* in0 = (t == 0) ? xb : h0b[prev];
    const bf16_t* w0  = (t == 0) ? wb_ih0 : wb_sum0;
    lstm_layer_wmma<<<gridBig, blk, 0, stream>>>(
        in0, w0, HID, (const bf16_t*)nullptr, (const bf16_t*)nullptr, 0, 0,
        bias0, c0, h0b[cur], HID, (float*)nullptr, t, HID);

    // Layer 1 (both paths K=512 -> both TDM-staged)
    const bf16_t* a2_1 = (t == 0) ? (const bf16_t*)nullptr : h1b[prev];
    lstm_layer_wmma<<<gridBig, blk, 0, stream>>>(
        h0b[cur], wb_ih1, HID, a2_1, wb_hh1, HID, 1,
        bias1, c1, h1b[cur], HID, (float*)nullptr, t, HID);

    // Layer 2 (padded hidden path K=96 via global; writes fp32 out[b,t,:])
    const bf16_t* a2_2 = (t == 0) ? (const bf16_t*)nullptr : h2b[prev];
    lstm_layer_wmma<<<gridOut, blk, 0, stream>>>(
        h1b[cur], wb_ih2, HID, a2_2, wb_hh2p, OUTP, 0,
        bias2, c2, h2b[cur], OUTP, out, t, OUTD);

    cur ^= 1;
  }
}
